// GIN_56865366999318
// MI455X (gfx1250) — matrix operand order repaired and checked by reading, not compiled, not run on hardware
//
#include <hip/hip_runtime.h>
#include <hip/hip_bf16.h>

// ---------------------------------------------------------------------------
// GIN forward, MI455X (gfx1250, wave32).
//   agg1 = x + scatter_add(x[src] -> dst)            (f32 atomics, L2-resident)
//   h    = relu( relu(agg1@W1+b1) @ W2 + b2 )        (fused WMMA MLP, 16-row tiles)
//   agg2 = h + scatter_add(h[src] -> dst)
//   out  =       relu(agg2@W3+b3) @ W4 + b4          (fused WMMA MLP)
// ---------------------------------------------------------------------------

typedef _Float16 v16h __attribute__((ext_vector_type(16)));
typedef _Float16 h8   __attribute__((ext_vector_type(8)));
typedef float    v8f  __attribute__((ext_vector_type(8)));

union Afrag { v16h v; h8 p[2]; };

__device__ __forceinline__ v8f wmma_f16(v16h a, v16h b, v8f c) {
    // D = A(16x32 f16) * B(32x16 f16) + C(16x16 f32)
    return __builtin_amdgcn_wmma_f32_16x16x32_f16(
        /*neg_a=*/false, a, /*neg_b=*/false, b,
        /*c_mod=*/(short)0, c, /*reuse_a=*/false, /*reuse_b=*/false);
}

// ---- weight convert + transpose: W[K][N] f32 -> Wt[N][K] f16 ---------------
__global__ __launch_bounds__(256) void gin_wcvt(const float* __restrict__ W,
                                                _Float16* __restrict__ Wt,
                                                int K, int Nout) {
    int t = blockIdx.x * 256 + threadIdx.x;
    if (t >= K * Nout) return;
    int k = t / Nout, n = t % Nout;
    Wt[n * K + k] = (_Float16)W[t];
}

// ---- vectorized f32 copy (agg init) ---------------------------------------
__global__ __launch_bounds__(256) void gin_copy4(const float4* __restrict__ s,
                                                 float4* __restrict__ d, int n4) {
    int i = blockIdx.x * 256 + threadIdx.x;
    if (i < n4) d[i] = s[i];
}

// ---- edge scatter-add: agg[dst][c] += feat[src][c], C = 1<<shift ----------
__global__ __launch_bounds__(256) void gin_scatter(const float* __restrict__ feat,
                                                   const long long* __restrict__ ei,
                                                   float* __restrict__ agg,
                                                   long long E_, int shift) {
    long long t = (long long)blockIdx.x * 256 + threadIdx.x;
    if (t >= (E_ << shift)) return;
    long long e = t >> shift;
    int c = (int)(t & ((1LL << shift) - 1));
    long long s = ei[e];          // src row
    long long d = ei[E_ + e];     // dst row
    atomicAdd(agg + (d << shift) + c, feat[(s << shift) + c]);
}

// ---- fused 2-layer MLP on 16-row node tiles, one wave per tile ------------
// in:[ntiles*16][KIN] f32, wtA:[NHID][KIN] f16, wtB:[NOUT][NHID] f16
// out (and optionally dup) : [ntiles*16][NOUT] f32
template<int KIN, int NHID, int NOUT, bool RELU_OUT, bool DUP>
__global__ __launch_bounds__(256) void gin_mlp16(
    const float* __restrict__ in,
    const _Float16* __restrict__ wtA, const float* __restrict__ bA,
    const _Float16* __restrict__ wtB, const float* __restrict__ bB,
    float* __restrict__ out, float* __restrict__ dup, int ntiles)
{
    __shared__ alignas(32) _Float16 sA[8][16 * KIN];   // staged input tile (f16)
    __shared__ alignas(32) _Float16 sT[8][16 * NHID];  // hidden activations (f16)

    const int lane = threadIdx.x & 31;
    const int wv   = threadIdx.x >> 5;
    const int tile = blockIdx.x * 8 + wv;
    if (tile >= ntiles) return;                        // wave-uniform: EXEC stays all-1s

    _Float16* A = sA[wv];
    _Float16* T = sT[wv];

    // Stage 16 x KIN f32 rows -> f16 LDS, float4-vectorized.
    const float4* src4 = reinterpret_cast<const float4*>(in + (size_t)tile * (16 * KIN));
    #pragma unroll
    for (int i = 0; i < (16 * KIN / 4) / 32; ++i) {
        const int idx = lane + 32 * i;
        const float4 vv = src4[idx];
        A[idx * 4 + 0] = (_Float16)vv.x;
        A[idx * 4 + 1] = (_Float16)vv.y;
        A[idx * 4 + 2] = (_Float16)vv.z;
        A[idx * 4 + 3] = (_Float16)vv.w;
    }
    __builtin_prefetch(wtB, 0, 0);                     // warm GEMM2 weights in L2/L0
    asm volatile("s_wait_dscnt 0" ::: "memory");       // LDS stores visible wave-wide

    const int lm = lane & 15;      // A row / BCD column within tile
    const int hi = lane >> 4;      // lane-half selector
    const int mb = hi << 3;        // C/D row base for this lane half

    // ---- GEMM1: [16xKIN] @ [KINxNHID] + bA, ReLU -> T (f16, LDS) ----
    #pragma unroll
    for (int nt = 0; nt < NHID / 16; ++nt) {
        const int nc = nt * 16 + lm;
        v8f acc{};
        const float bv = bA[nc];
        #pragma unroll
        for (int r = 0; r < 8; ++r) acc[r] = bv;
        #pragma unroll
        for (int ks = 0; ks < KIN; ks += 32) {
            Afrag a;   // A layout matches B: lanes0-15 K0-15, lanes16-31 K16-31
            a.p[0] = *reinterpret_cast<const h8*>(&A[lm * KIN + ks + 16 * hi]);
            a.p[1] = *reinterpret_cast<const h8*>(&A[lm * KIN + ks + 16 * hi + 8]);
            // B layout: lane = column, lanes0-15 K0-15 / lanes16-31 K16-31 -> one 32B load of Wt
            const v16h b = *reinterpret_cast<const v16h*>(&wtA[(size_t)nc * KIN + ks + 16 * hi]);
            acc = wmma_f16(a.v, b, acc);
        }
        #pragma unroll
        for (int r = 0; r < 8; ++r) {
            float t = acc[r] > 0.f ? acc[r] : 0.f;     // hidden ReLU
            T[(mb + r) * NHID + nc] = (_Float16)t;     // C-layout -> row-major via LDS
        }
    }
    asm volatile("s_wait_dscnt 0" ::: "memory");

    // ---- GEMM2: [16xNHID] @ [NHIDxNOUT] + bB (+ReLU) -> out ----
    #pragma unroll
    for (int nt = 0; nt < NOUT / 16; ++nt) {
        const int nc = nt * 16 + lm;
        v8f acc{};
        const float bv = bB[nc];
        #pragma unroll
        for (int r = 0; r < 8; ++r) acc[r] = bv;
        #pragma unroll
        for (int ks = 0; ks < NHID; ks += 32) {
            Afrag a;
            a.p[0] = *reinterpret_cast<const h8*>(&T[lm * NHID + ks + 16 * hi]);
            a.p[1] = *reinterpret_cast<const h8*>(&T[lm * NHID + ks + 16 * hi + 8]);
            const v16h b = *reinterpret_cast<const v16h*>(&wtB[(size_t)nc * NHID + ks + 16 * hi]);
            acc = wmma_f16(a.v, b, acc);
        }
        #pragma unroll
        for (int r = 0; r < 8; ++r) {
            float t = acc[r];
            if (RELU_OUT) t = t > 0.f ? t : 0.f;
            const size_t o = (size_t)(tile * 16 + mb + r) * NOUT + nc;
            out[o] = t;
            if (DUP) dup[o] = t;                       // agg2 init fused with h store
        }
    }
}

// ---------------------------------------------------------------------------
extern "C" void kernel_launch(void* const* d_in, const int* in_sizes, int n_in,
                              void* d_out, int out_size, void* d_ws, size_t ws_size,
                              hipStream_t stream) {
    (void)n_in; (void)out_size; (void)ws_size;

    const float*     x  = (const float*)d_in[0];
    const long long* ei = (const long long*)d_in[1];      // int64 [2][E]
    const float* W1 = (const float*)d_in[2]; const float* b1 = (const float*)d_in[3];
    const float* W2 = (const float*)d_in[4]; const float* b2 = (const float*)d_in[5];
    const float* W3 = (const float*)d_in[6]; const float* b3 = (const float*)d_in[7];
    const float* W4 = (const float*)d_in[8]; const float* b4 = (const float*)d_in[9];

    const int       Nn = in_sizes[0] / 32;                // 100000
    const long long E_ = in_sizes[1] / 2;                 // 1600000
    const int   ntiles = Nn / 16;                         // 6250 (N % 16 == 0)

    // workspace carve-up
    float* agg1 = (float*)d_ws;                           // [N][32]
    float* hbuf = agg1 + (size_t)Nn * 32;                 // [N][64]
    float* agg2 = hbuf + (size_t)Nn * 64;                 // [N][64]
    _Float16* wt1 = (_Float16*)(agg2 + (size_t)Nn * 64);  // [64][32]
    _Float16* wt2 = wt1 + 64 * 32;                        // [64][64]
    _Float16* wt3 = wt2 + 64 * 64;                        // [64][64]
    _Float16* wt4 = wt3 + 64 * 64;                        // [32][64]

    // 1) weights: f32 [K][N] -> f16 [N][K]
    gin_wcvt<<<(32 * 64 + 255) / 256, 256, 0, stream>>>(W1, wt1, 32, 64);
    gin_wcvt<<<(64 * 64 + 255) / 256, 256, 0, stream>>>(W2, wt2, 64, 64);
    gin_wcvt<<<(64 * 64 + 255) / 256, 256, 0, stream>>>(W3, wt3, 64, 64);
    gin_wcvt<<<(64 * 32 + 255) / 256, 256, 0, stream>>>(W4, wt4, 64, 32);

    // 2) agg1 = x  (self term, eps = 0)
    {
        int n4 = Nn * 32 / 4;
        gin_copy4<<<(n4 + 255) / 256, 256, 0, stream>>>((const float4*)x, (float4*)agg1, n4);
    }

    // 3) agg1 += scatter(x[src] -> dst), 32 channels
    {
        long long total = E_ << 5;
        gin_scatter<<<(int)((total + 255) / 256), 256, 0, stream>>>(x, ei, agg1, E_, 5);
    }

    // 4) h = relu(MLP1(agg1)); agg2 = h (fused dup-store)
    gin_mlp16<32, 64, 64, true, true>
        <<<(ntiles + 7) / 8, 256, 0, stream>>>(agg1, wt1, b1, wt2, b2, hbuf, agg2, ntiles);

    // 5) agg2 += scatter(h[src] -> dst), 64 channels
    {
        long long total = E_ << 6;
        gin_scatter<<<(int)((total + 255) / 256), 256, 0, stream>>>(hbuf, ei, agg2, E_, 6);
    }

    // 6) out = MLP2(agg2), no final relu
    gin_mlp16<64, 64, 32, false, false>
        <<<(ntiles + 7) / 8, 256, 0, stream>>>(agg2, wt3, b3, wt4, b4, (float*)d_out, nullptr, ntiles);
}
